// RNN_16509854286418
// MI455X (gfx1250) — compile-verified, matrix operand order checked by reading, not run on hardware
//
#include <hip/hip_runtime.h>
#include <hip/hip_bf16.h>

typedef __attribute__((ext_vector_type(16))) __bf16 v16bf;
typedef __attribute__((ext_vector_type(8)))  __bf16 v8bf;
typedef __attribute__((ext_vector_type(8)))  float  v8f;
typedef __attribute__((ext_vector_type(4)))  float  v4f;

#define B_    4
#define T_    8192
#define DIN   512
#define H_    512
#define DOUT  512
#define M_    (B_ * T_)        // 32768
#define CHUNK 128              // time-chunk == M tile
#define NCHPB (T_ / CHUNK)     // 64 chunks per batch
#define NCH   (M_ / CHUNK)     // 256 global chunks
#define MT    128
#define NT    128

// fp32 -> bf16, round-to-nearest-even
__device__ __forceinline__ __bf16 f2bf(float f) {
  union { float f; unsigned u; } x; x.f = f;
  unsigned r = x.u + 0x7FFFu + ((x.u >> 16) & 1u);
  union { unsigned short s; __bf16 b; } y; y.s = (unsigned short)(r >> 16);
  return y.b;
}

// ---------------- prep: W^T -> bf16, a-power table -------------------------
__global__ __launch_bounds__(256) void k_prep(
    const float* __restrict__ Win, const float* __restrict__ Wout,
    const float* __restrict__ a,
    __bf16* __restrict__ WtIn, __bf16* __restrict__ WtOut,
    float* __restrict__ apow) {
  int idx = blockIdx.x * 256 + threadIdx.x;
  if (idx < DIN * H_) {
    int n = idx >> 9, k = idx & 511;
    WtIn[idx]  = f2bf(Win[(size_t)k * H_  + n]);   // WtIn[n][k] = Win[k][n]
    WtOut[idx] = f2bf(Wout[(size_t)k * DOUT + n]); // WtOut[n][k] = Wout[k][n]
  }
  if (idx < H_) {
    float av = a[idx], p = 1.f;
    for (int j = 0; j <= CHUNK; ++j) { apow[(size_t)j * H_ + idx] = p; p *= av; }
  }
}

// ---------------- GEMM1 (u = x @ Win + b_in) + fused local scan ------------
__global__ __launch_bounds__(256) void k_gemm1_scan(
    const float* __restrict__ x, const __bf16* __restrict__ wt,
    const float* __restrict__ b_in, const float* __restrict__ a,
    float* __restrict__ s_out, float* __restrict__ carry) {
  __shared__ float lds[MT * NT];                 // 64 KB tile
  const int tileN = blockIdx.x & 3;              // H/NT = 4
  const int tileM = blockIdx.x >> 2;             // == global chunk id
  const int wave  = threadIdx.x >> 5;
  const int lane  = threadIdx.x & 31;
  const int lhalf = lane & 15;
  const int koff  = (lane < 16) ? 0 : 8;
  const int M0 = tileM * MT + wave * 16;
  const int N0 = tileN * NT;

  v8f acc[8];
  #pragma unroll
  for (int i = 0; i < 8; ++i) acc[i] = (v8f)0.0f;

  const float* arow = x + (size_t)(M0 + lhalf) * DIN;

  #pragma unroll 1
  for (int kk = 0; kk < DIN; kk += 32) {
    v16bf afrag;
    {
      const float* p = arow + kk + koff;
      v4f f0 = *(const v4f*)(p);
      v4f f1 = *(const v4f*)(p + 4);
      v4f f2 = *(const v4f*)(p + 16);
      v4f f3 = *(const v4f*)(p + 20);
      #pragma unroll
      for (int i = 0; i < 4; ++i) {
        afrag[i]      = f2bf(f0[i]);  afrag[4 + i]  = f2bf(f1[i]);
        afrag[8 + i]  = f2bf(f2[i]);  afrag[12 + i] = f2bf(f3[i]);
      }
    }
    #pragma unroll
    for (int nt = 0; nt < 8; ++nt) {
      const __bf16* bp = wt + (size_t)(N0 + nt * 16 + lhalf) * DIN + kk + koff;
      v8bf b0 = *(const v8bf*)(bp);
      v8bf b1 = *(const v8bf*)(bp + 16);
      v16bf bfrag;
      #pragma unroll
      for (int i = 0; i < 8; ++i) { bfrag[i] = b0[i]; bfrag[8 + i] = b1[i]; }
      acc[nt] = __builtin_amdgcn_wmma_f32_16x16x32_bf16(
          false, afrag, false, bfrag, (short)0, acc[nt], false, false);
    }
  }

  // epilogue: acc (+bias) -> LDS   (D layout: vgpr r -> M=r / r+8 per lane half)
  const int moff = wave * 16 + ((lane < 16) ? 0 : 8);
  #pragma unroll
  for (int nt = 0; nt < 8; ++nt) {
    int ncol = nt * 16 + lhalf;
    float bias = b_in[N0 + ncol];
    #pragma unroll
    for (int r = 0; r < 8; ++r) lds[(moff + r) * NT + ncol] = acc[nt][r] + bias;
  }
  __syncthreads();

  // local scan along time (M) for 128 hidden columns of this tile
  if (threadIdx.x < NT) {
    int c = threadIdx.x;
    float ac = a[N0 + c];
    float s = 0.f;
    #pragma unroll 4
    for (int i = 0; i < MT; ++i) { s = ac * s + lds[i * NT + c]; lds[i * NT + c] = s; }
    carry[(size_t)tileM * H_ + N0 + c] = s;   // chunk-final local value L_j
  }
  __syncthreads();

  // coalesced b128 tile store: s_out[m][h]
  #pragma unroll
  for (int j = 0; j < (MT * NT / 4) / 256; ++j) {
    int idx4 = j * 256 + threadIdx.x;          // over 4096 float4
    int ml = idx4 >> 5, n4 = (idx4 & 31) * 4;
    *(v4f*)(s_out + (size_t)(tileM * MT + ml) * H_ + N0 + n4) =
        *(const v4f*)(&lds[ml * NT + n4]);
  }
}

// ---------------- carry scan over chunks (sequential, tiny) ----------------
__global__ __launch_bounds__(256) void k_cscan(
    float* __restrict__ carry, const float* __restrict__ apow) {
  int tid = blockIdx.x * 256 + threadIdx.x;    // 0..2047 = (b, h)
  if (tid >= B_ * H_) return;
  int b = tid >> 9, hch = tid & 511;
  float aC = apow[(size_t)CHUNK * H_ + hch];   // a^128
  float c = 0.f;
  #pragma unroll 1
  for (int j = 0; j < NCHPB; ++j) {
    size_t idx = (size_t)(b * NCHPB + j) * H_ + hch;
    float L = carry[idx];
    carry[idx] = c;                            // exclusive prefix (carry-in)
    c = aC * c + L;
  }
}

// ------- GEMM2 (out = h @ Wout + b_out), h = s + a^{i+1} * carry fused -----
__global__ __launch_bounds__(256) void k_gemm2(
    const float* __restrict__ s_in, const __bf16* __restrict__ wt,
    const float* __restrict__ carry, const float* __restrict__ apow,
    const float* __restrict__ b_out, float* __restrict__ out) {
  const int tileN = blockIdx.x & 3;
  const int tileM = blockIdx.x >> 2;
  const int wave  = threadIdx.x >> 5;
  const int lane  = threadIdx.x & 31;
  const int lhalf = lane & 15;
  const int koff  = (lane < 16) ? 0 : 8;
  const int M0 = tileM * MT + wave * 16;
  const int N0 = tileN * NT;

  v8f acc[8];
  #pragma unroll
  for (int i = 0; i < 8; ++i) acc[i] = (v8f)0.0f;

  const int m  = M0 + lhalf;
  const int i1 = (m & (CHUNK - 1)) + 1;        // local time index + 1
  const float* srow = s_in  + (size_t)m * H_;
  const float* prow = apow  + (size_t)i1 * H_;
  const float* crow = carry + (size_t)(m >> 7) * H_;

  #pragma unroll 1
  for (int kk = 0; kk < H_; kk += 32) {
    v16bf afrag;
    {
      int o = kk + koff;
      v4f s0 = *(const v4f*)(srow + o),      s1 = *(const v4f*)(srow + o + 4);
      v4f s2 = *(const v4f*)(srow + o + 16), s3 = *(const v4f*)(srow + o + 20);
      v4f p0 = *(const v4f*)(prow + o),      p1 = *(const v4f*)(prow + o + 4);
      v4f p2 = *(const v4f*)(prow + o + 16), p3 = *(const v4f*)(prow + o + 20);
      v4f c0 = *(const v4f*)(crow + o),      c1 = *(const v4f*)(crow + o + 4);
      v4f c2 = *(const v4f*)(crow + o + 16), c3 = *(const v4f*)(crow + o + 20);
      #pragma unroll
      for (int i = 0; i < 4; ++i) {
        afrag[i]      = f2bf(s0[i] + p0[i] * c0[i]);
        afrag[4 + i]  = f2bf(s1[i] + p1[i] * c1[i]);
        afrag[8 + i]  = f2bf(s2[i] + p2[i] * c2[i]);
        afrag[12 + i] = f2bf(s3[i] + p3[i] * c3[i]);
      }
    }
    #pragma unroll
    for (int nt = 0; nt < 8; ++nt) {
      const __bf16* bp = wt + (size_t)(N0 + nt * 16 + lhalf) * H_ + kk + koff;
      v8bf b0 = *(const v8bf*)(bp);
      v8bf b1 = *(const v8bf*)(bp + 16);
      v16bf bfrag;
      #pragma unroll
      for (int i = 0; i < 8; ++i) { bfrag[i] = b0[i]; bfrag[8 + i] = b1[i]; }
      acc[nt] = __builtin_amdgcn_wmma_f32_16x16x32_bf16(
          false, afrag, false, bfrag, (short)0, acc[nt], false, false);
    }
  }

  const int mbase = M0 + ((lane < 16) ? 0 : 8);
  #pragma unroll
  for (int nt = 0; nt < 8; ++nt) {
    int n = N0 + nt * 16 + lhalf;
    float bias = b_out[n];
    #pragma unroll
    for (int r = 0; r < 8; ++r)
      out[(size_t)(mbase + r) * DOUT + n] = acc[nt][r] + bias;
  }
}

// ---------------------------------------------------------------------------
extern "C" void kernel_launch(void* const* d_in, const int* in_sizes, int n_in,
                              void* d_out, int out_size, void* d_ws, size_t ws_size,
                              hipStream_t stream) {
  const float* x     = (const float*)d_in[0];
  const float* a     = (const float*)d_in[1];
  const float* W_in  = (const float*)d_in[2];
  const float* b_in  = (const float*)d_in[3];
  const float* W_out = (const float*)d_in[4];
  const float* b_out = (const float*)d_in[5];
  float* out = (float*)d_out;

  char* ws = (char*)d_ws;
  float*  ws_s     = (float*)(ws);                                  // 64 MiB
  float*  ws_carry = (float*)(ws + 67108864);                       // 512 KiB
  __bf16* ws_wtin  = (__bf16*)(ws + 67108864 + 524288);             // 512 KiB
  __bf16* ws_wtout = (__bf16*)(ws + 67108864 + 2 * 524288);         // 512 KiB
  float*  ws_apow  = (float*)(ws + 67108864 + 3 * 524288);          // 258 KiB

  k_prep<<<dim3((DIN * H_) / 256), dim3(256), 0, stream>>>(
      W_in, W_out, a, ws_wtin, ws_wtout, ws_apow);

  k_gemm1_scan<<<dim3((M_ / MT) * (H_ / NT)), dim3(256), 0, stream>>>(
      x, ws_wtin, b_in, a, ws_s, ws_carry);

  k_cscan<<<dim3((B_ * H_) / 256), dim3(256), 0, stream>>>(ws_carry, ws_apow);

  k_gemm2<<<dim3((M_ / MT) * (DOUT / NT)), dim3(256), 0, stream>>>(
      ws_s, ws_wtout, ws_carry, ws_apow, b_out, out);
}
